// MoGCN_66726611910977
// MI455X (gfx1250) — matrix operand order repaired and checked by reading, not compiled, not run on hardware
//
#include <hip/hip_runtime.h>
#include <cstddef>

typedef _Float16 half_t;
typedef half_t v16h __attribute__((ext_vector_type(16)));
typedef half_t v8h  __attribute__((ext_vector_type(8)));
typedef float  v8f  __attribute__((ext_vector_type(8)));

// ---------------------------------------------------------------------------
// Elementwise / graph-prep kernels
// ---------------------------------------------------------------------------
__global__ void mogcn_zero_f32(float* __restrict__ p, long n) {
    long i = (long)blockIdx.x * blockDim.x + threadIdx.x;
    if (i < n) p[i] = 0.0f;
}

__global__ void mogcn_deg_accum(const int* __restrict__ dst, float* __restrict__ deg, int E) {
    int e = blockIdx.x * blockDim.x + threadIdx.x;
    if (e < E) atomicAdd(&deg[dst[e]], 1.0f);
}

__global__ void mogcn_dinv(float* __restrict__ dinv, int N) {
    int i = blockIdx.x * blockDim.x + threadIdx.x;
    if (i < N) dinv[i] = rsqrtf(dinv[i] + 1.0f);   // +1 for self-loop; deg >= 1 always
}

__global__ void mogcn_norm(const int* __restrict__ src, const int* __restrict__ dst,
                           const float* __restrict__ dinv, float* __restrict__ norm, int E) {
    int e = blockIdx.x * blockDim.x + threadIdx.x;
    if (e < E) norm[e] = dinv[src[e]] * dinv[dst[e]];
}

// agg[i][j] = h[i][j] * dinv[i]^2 + b[j]   (self-loop + bias fused into accumulator init)
__global__ void mogcn_agg_init(const float* __restrict__ h, const float* __restrict__ dinv,
                               const float* __restrict__ b, float* __restrict__ agg, int N) {
    int t = blockIdx.x * blockDim.x + threadIdx.x;
    int node = t >> 4;
    if (node >= N) return;
    int j = (t & 15) * 4;
    float di = dinv[node];
    float w = di * di;
    const float4 v = *(const float4*)(h + (size_t)node * 64 + j);
    float4 r;
    r.x = v.x * w + b[j + 0];
    r.y = v.y * w + b[j + 1];
    r.z = v.z * w + b[j + 2];
    r.w = v.w * w + b[j + 3];
    *(float4*)(agg + (size_t)node * 64 + j) = r;
}

// agg[dst][j] += h[src][j] * norm[e]
__global__ void mogcn_scatter(const int* __restrict__ src, const int* __restrict__ dst,
                              const float* __restrict__ norm, const float* __restrict__ h,
                              float* __restrict__ agg, int E) {
    int t = blockIdx.x * blockDim.x + threadIdx.x;
    int e = t >> 4;
    if (e >= E) return;
    int j = (t & 15) * 4;
    int s = src[e], d = dst[e];
    float nm = norm[e];
    const float4 v = *(const float4*)(h + (size_t)s * 64 + j);
    float* o = agg + (size_t)d * 64 + j;
    atomicAdd(o + 0, v.x * nm);
    atomicAdd(o + 1, v.y * nm);
    atomicAdd(o + 2, v.z * nm);
    atomicAdd(o + 3, v.w * nm);
}

__global__ void mogcn_elu(const float* __restrict__ in, float* __restrict__ out, long n) {
    long i = (long)blockIdx.x * blockDim.x + threadIdx.x;
    if (i < n) {
        float v = in[i];
        out[i] = v > 0.0f ? v : expm1f(v);
    }
}

// ---------------------------------------------------------------------------
// WMMA GEMM: H[N,64] = X[N,K] @ W[K,64], f32 in/out, f16 multiply, f32 accumulate
// Block: 256 threads = 8 waves, 128 rows. Wave: 16 rows x 64 cols (4 accumulators).
// ---------------------------------------------------------------------------
__device__ inline v16h mogcn_load_b16(const half_t* p) {   // 16 contiguous f16 (two 16B LDS loads)
    v8h lo = *(const v8h*)(p);
    v8h hi = *(const v8h*)(p + 8);
    return __builtin_shufflevector(lo, hi, 0,1,2,3,4,5,6,7,8,9,10,11,12,13,14,15);
}

__device__ inline void mogcn_cvt8(const float* __restrict__ p, v16h& a, int off) {
    float4 f0 = *(const float4*)(p);
    float4 f1 = *(const float4*)(p + 4);
    a[off + 0] = (half_t)f0.x; a[off + 1] = (half_t)f0.y;
    a[off + 2] = (half_t)f0.z; a[off + 3] = (half_t)f0.w;
    a[off + 4] = (half_t)f1.x; a[off + 5] = (half_t)f1.y;
    a[off + 6] = (half_t)f1.z; a[off + 7] = (half_t)f1.w;
}

template<int K>
__global__ __launch_bounds__(256)
void mogcn_gemm(const float* __restrict__ X, const float* __restrict__ W,
                float* __restrict__ Hout, int nrows) {
    constexpr int KP = K + 8;                 // padded LDS stride (f16 units) to dodge bank conflicts
    __shared__ half_t lB[64 * KP];            // W transposed: lB[n*KP + k] = W[k][n]

    // Cooperative convert + transpose W -> LDS (coalesced global reads)
    for (int idx = threadIdx.x; idx < K * 64; idx += 256) {
        int k = idx / 64, n = idx % 64;
        lB[n * KP + k] = (half_t)W[idx];
    }
    __syncthreads();

    const int wave = threadIdx.x >> 5;
    const int lane = threadIdx.x & 31;
    const int m    = lane & 15;               // M (A) / N (B,D) index within 16-tile
    const int hi   = lane >> 4;               // K-half selector
    const int rowBase = blockIdx.x * 128 + wave * 16;
    int row = rowBase + m;
    if (row > nrows - 1) row = nrows - 1;     // clamp loads; stores guarded later
    const float* arow = X + (size_t)row * K;

    v8f acc0 = {}, acc1 = {}, acc2 = {}, acc3 = {};

    for (int k0 = 0; k0 < K; k0 += 32) {
        // A operand (16x32 f16): lane(hi=0): K = k0+[0..15]
        //                        lane(hi=1): K = k0+[16..31]
        v16h a;
        const int ka = k0 + hi * 16;
        mogcn_cvt8(arow + ka,      a, 0);
        mogcn_cvt8(arow + ka + 8,  a, 8);

        // B operand (32x16 f16): lane n = m, K = k0 + hi*16 + [0..15], contiguous in LDS
        const int kb = k0 + hi * 16;
        v16h b0 = mogcn_load_b16(&lB[(0 * 16 + m) * KP + kb]);
        v16h b1 = mogcn_load_b16(&lB[(1 * 16 + m) * KP + kb]);
        v16h b2 = mogcn_load_b16(&lB[(2 * 16 + m) * KP + kb]);
        v16h b3 = mogcn_load_b16(&lB[(3 * 16 + m) * KP + kb]);

        acc0 = __builtin_amdgcn_wmma_f32_16x16x32_f16(false, a, false, b0, (short)0, acc0, false, false);
        acc1 = __builtin_amdgcn_wmma_f32_16x16x32_f16(false, a, false, b1, (short)0, acc1, false, false);
        acc2 = __builtin_amdgcn_wmma_f32_16x16x32_f16(false, a, false, b2, (short)0, acc2, false, false);
        acc3 = __builtin_amdgcn_wmma_f32_16x16x32_f16(false, a, false, b3, (short)0, acc3, false, false);
    }

    // D layout: lane col n = m; rows = hi*8 + r
    const int srow0 = rowBase + hi * 8;
    #pragma unroll
    for (int r = 0; r < 8; ++r) {
        int rr = srow0 + r;
        if (rr < nrows) {
            float* orow = Hout + (size_t)rr * 64 + m;
            orow[0]  = acc0[r];
            orow[16] = acc1[r];
            orow[32] = acc2[r];
            orow[48] = acc3[r];
        }
    }
}

// ---------------------------------------------------------------------------
// Final classifier: out[n,c] = [x1|x2][n,:] @ Wfc + bfc   (128x4, tiny -> VALU)
// ---------------------------------------------------------------------------
__global__ __launch_bounds__(256)
void mogcn_final(const float* __restrict__ x1, const float* __restrict__ x2,
                 const float* __restrict__ Wfc, const float* __restrict__ bfc,
                 float* __restrict__ out, int N) {
    __shared__ float w[128 * 4];
    __shared__ float bb[4];
    for (int i = threadIdx.x; i < 512; i += 256) w[i] = Wfc[i];
    if (threadIdx.x < 4) bb[threadIdx.x] = bfc[threadIdx.x];
    __syncthreads();

    int n = blockIdx.x * blockDim.x + threadIdx.x;
    if (n >= N) return;
    float a0 = bb[0], a1 = bb[1], a2 = bb[2], a3 = bb[3];
    const float* r1 = x1 + (size_t)n * 64;
    const float* r2 = x2 + (size_t)n * 64;
    #pragma unroll 4
    for (int k = 0; k < 64; ++k) {
        float v = r1[k];
        a0 += v * w[k * 4 + 0]; a1 += v * w[k * 4 + 1];
        a2 += v * w[k * 4 + 2]; a3 += v * w[k * 4 + 3];
    }
    #pragma unroll 4
    for (int k = 0; k < 64; ++k) {
        float v = r2[k];
        a0 += v * w[(64 + k) * 4 + 0]; a1 += v * w[(64 + k) * 4 + 1];
        a2 += v * w[(64 + k) * 4 + 2]; a3 += v * w[(64 + k) * 4 + 3];
    }
    float* o = out + (size_t)n * 4;
    o[0] = a0; o[1] = a1; o[2] = a2; o[3] = a3;
}

// ---------------------------------------------------------------------------
// Host orchestration
// ---------------------------------------------------------------------------
static inline int cdiv_l(long a, long b) { return (int)((a + b - 1) / b); }

extern "C" void kernel_launch(void* const* d_in, const int* in_sizes, int n_in,
                              void* d_out, int out_size, void* d_ws, size_t ws_size,
                              hipStream_t stream) {
    const float* x1in = (const float*)d_in[0];
    const float* x2in = (const float*)d_in[1];
    const int*   eidx = (const int*)d_in[2];
    const float* W1a = (const float*)d_in[3];  const float* b1a = (const float*)d_in[4];
    const float* W2a = (const float*)d_in[5];  const float* b2a = (const float*)d_in[6];
    const float* W1b = (const float*)d_in[7];  const float* b1b = (const float*)d_in[8];
    const float* W2b = (const float*)d_in[9];  const float* b2b = (const float*)d_in[10];
    const float* Wfc = (const float*)d_in[11]; const float* bfc = (const float*)d_in[12];

    const int D1 = 512;
    const int N  = in_sizes[0] / D1;
    const int E  = in_sizes[2] / 2;
    const int* srcI = eidx;
    const int* dstI = eidx + E;

    // Workspace layout (f32): dinv[N] | norm[E] | hbuf[N*64] | aggb[N*64] | x1[N*64]
    float* ws   = (float*)d_ws;
    float* dinv = ws;
    float* norm = dinv + N;
    float* hbuf = norm + E;
    float* aggb = hbuf + (size_t)N * 64;
    float* x1b  = aggb + (size_t)N * 64;

    const long NE = (long)N * 64;

    // --- degrees & edge norms ---
    mogcn_zero_f32<<<cdiv_l(N, 256), 256, 0, stream>>>(dinv, N);
    mogcn_deg_accum<<<cdiv_l(E, 256), 256, 0, stream>>>(dstI, dinv, E);
    mogcn_dinv<<<cdiv_l(N, 256), 256, 0, stream>>>(dinv, N);
    mogcn_norm<<<cdiv_l(E, 256), 256, 0, stream>>>(srcI, dstI, dinv, norm, E);

    const int gemmGrid = cdiv_l(N, 128);
    const int nodeV    = cdiv_l((long)N * 16, 256);
    const int edgeV    = cdiv_l((long)E * 16, 256);
    const int elemG    = cdiv_l(NE, 256);

    // --- branch 1 (omic1: 512 -> 64 -> 64), result into x1b ---
    mogcn_gemm<512><<<gemmGrid, 256, 0, stream>>>(x1in, W1a, hbuf, N);
    mogcn_agg_init<<<nodeV, 256, 0, stream>>>(hbuf, dinv, b1a, aggb, N);
    mogcn_scatter<<<edgeV, 256, 0, stream>>>(srcI, dstI, norm, hbuf, aggb, E);
    mogcn_elu<<<elemG, 256, 0, stream>>>(aggb, aggb, NE);

    mogcn_gemm<64><<<gemmGrid, 256, 0, stream>>>(aggb, W2a, hbuf, N);
    mogcn_agg_init<<<nodeV, 256, 0, stream>>>(hbuf, dinv, b2a, aggb, N);
    mogcn_scatter<<<edgeV, 256, 0, stream>>>(srcI, dstI, norm, hbuf, aggb, E);
    mogcn_elu<<<elemG, 256, 0, stream>>>(aggb, x1b, NE);

    // --- branch 2 (omic2: 256 -> 64 -> 64), result ends in aggb (in place) ---
    mogcn_gemm<256><<<gemmGrid, 256, 0, stream>>>(x2in, W1b, hbuf, N);
    mogcn_agg_init<<<nodeV, 256, 0, stream>>>(hbuf, dinv, b1b, aggb, N);
    mogcn_scatter<<<edgeV, 256, 0, stream>>>(srcI, dstI, norm, hbuf, aggb, E);
    mogcn_elu<<<elemG, 256, 0, stream>>>(aggb, aggb, NE);

    mogcn_gemm<64><<<gemmGrid, 256, 0, stream>>>(aggb, W2b, hbuf, N);
    mogcn_agg_init<<<nodeV, 256, 0, stream>>>(hbuf, dinv, b2b, aggb, N);
    mogcn_scatter<<<edgeV, 256, 0, stream>>>(srcI, dstI, norm, hbuf, aggb, E);
    mogcn_elu<<<elemG, 256, 0, stream>>>(aggb, aggb, NE);

    // --- final classifier ---
    mogcn_final<<<cdiv_l(N, 256), 256, 0, stream>>>(x1b, aggb, Wfc, bfc, (float*)d_out, N);
}
